// TransformerEncoderLayerSpatial_987842478856
// MI455X (gfx1250) — matrix operand order repaired and checked by reading, not compiled, not run on hardware
//
#include <hip/hip_runtime.h>

typedef __attribute__((ext_vector_type(16))) _Float16 v16h;
typedef __attribute__((ext_vector_type(8)))  float    v8f;
typedef __attribute__((ext_vector_type(4)))  float    f32x4;
typedef __attribute__((ext_vector_type(2)))  unsigned int u32x2;
typedef __attribute__((ext_vector_type(4)))  unsigned int u32x4;
typedef __attribute__((ext_vector_type(8)))  int      i32x8;
typedef __attribute__((ext_vector_type(4)))  int      i32x4;

#define BZ   2
#define CDIM 256
#define NSEQ 4096
#define NH   8
#define HD   32
#define FF   1024
#define HPAD 68

#if defined(__has_builtin)
#if __has_builtin(__builtin_amdgcn_tensor_load_to_lds)
#define USE_TDM 1
#endif
#endif
#ifndef USE_TDM
#define USE_TDM 0
#endif

union AF {
  v16h v;
  _Float16 h[16];
  unsigned int d[8];
  u32x2 p[4];
  u32x4 q[2];
};

__device__ inline v8f v8f_zero() {
  v8f z;
#pragma unroll
  for (int i = 0; i < 8; ++i) z[i] = 0.f;
  return z;
}

__device__ inline v8f wmma16(const AF& a, const AF& b, v8f c) {
  return __builtin_amdgcn_wmma_f32_16x16x32_f16(false, a.v, false, b.v, (short)0, c, false, false);
}

// A-fragment (16x32 f16) from f32 row-major (+optional posemb row), b128 loads.
template <bool PE>
__device__ inline void load_a_f32(AF& A, const float* rowp, const float* pep, int k0, int half) {
  int klo = k0 + (half ? 16 : 0);
  int khi = klo + 8;
  f32x4 a0 = *(const f32x4*)(rowp + klo);
  f32x4 a1 = *(const f32x4*)(rowp + klo + 4);
  f32x4 b0 = *(const f32x4*)(rowp + khi);
  f32x4 b1 = *(const f32x4*)(rowp + khi + 4);
  if (PE) {
    a0 += *(const f32x4*)(pep + klo);
    a1 += *(const f32x4*)(pep + klo + 4);
    b0 += *(const f32x4*)(pep + khi);
    b1 += *(const f32x4*)(pep + khi + 4);
  }
#pragma unroll
  for (int t = 0; t < 4; ++t) {
    A.h[t] = (_Float16)a0[t];      A.h[4 + t] = (_Float16)a1[t];
    A.h[8 + t] = (_Float16)b0[t];  A.h[12 + t] = (_Float16)b1[t];
  }
}

// B-fragment (32x16 f16) from f16 row-major weight, 2x b128.
__device__ inline void load_b_f16(AF& B, const _Float16* wrow) {
  B.q[0] = *(const u32x4*)(wrow);
  B.q[1] = *(const u32x4*)(wrow + 8);
}

// B-fragment from f32 source (16 contiguous floats), cvt to f16.
__device__ inline void load_b_f32cvt(AF& B, const float* wrow) {
#pragma unroll
  for (int j = 0; j < 4; ++j) {
    f32x4 x = *(const f32x4*)(wrow + 4 * j);
#pragma unroll
    for (int t = 0; t < 4; ++t) B.h[4 * j + t] = (_Float16)x[t];
  }
}

// ---------------- small helper kernels ----------------

__global__ void cvt_f16_kernel(const float* __restrict__ src, _Float16* __restrict__ dst, int n) {
  int i = blockIdx.x * 256 + threadIdx.x;
  if (i < n) dst[i] = (_Float16)src[i];
}

// x[b,n,c] = src[b,c,i,j], n = i*64+j
__global__ void transpose_src_kernel(const float* __restrict__ src, float* __restrict__ x) {
  int e = blockIdx.x * 256 + threadIdx.x;
  if (e >= BZ * NSEQ * CDIM) return;
  int c = e & 255; int n = (e >> 8) & 4095; int b = e >> 20;
  x[e] = src[(((size_t)(b * CDIM + c)) << 12) + n];
}

// xrm[b,j,c] = mean_i x + posemb_row ; xcm[b,i,c] = mean_j x + posemb_col
__global__ void mean_pe_kernel(const float* __restrict__ x, const float* __restrict__ per,
                               const float* __restrict__ pec, float* __restrict__ xrm,
                               float* __restrict__ xcm) {
  int e = blockIdx.x * 256 + threadIdx.x;
  const int half_n = BZ * 64 * CDIM;
  if (e >= 2 * half_n) return;
  int sel = (e >= half_n);
  int r = e - sel * half_n;
  int c = r & 255; int p = (r >> 8) & 63; int b = r >> 14;
  const float* xb = x + ((size_t)b << 20);
  float s = 0.f;
  if (!sel) {
    for (int i = 0; i < 64; ++i) s += xb[(i * 64 + p) * 256 + c];
    xrm[r] = s * (1.f / 64.f) + per[r];
  } else {
    for (int j = 0; j < 64; ++j) s += xb[(p * 64 + j) * 256 + c];
    xcm[r] = s * (1.f / 64.f) + pec[r];
  }
}

// k_row[b,j,o] = xrm[b,j,:]·Wk_row[o,:] + bk_row[o]  (and col variant)
__global__ void kproj_kernel(const float* __restrict__ xrm, const float* __restrict__ xcm,
                             const float* __restrict__ Wkr, const float* __restrict__ bkr,
                             const float* __restrict__ Wkc, const float* __restrict__ bkc,
                             float* __restrict__ krow, float* __restrict__ kcol) {
  int e = blockIdx.x * 256 + threadIdx.x;
  const int half_n = BZ * 64 * CDIM;
  if (e >= 2 * half_n) return;
  int sel = (e >= half_n);
  int r = e - sel * half_n;
  int o = r & 255; int p = (r >> 8) & 63; int b = r >> 14;
  const float* in = (sel ? xcm : xrm) + ((size_t)(b * 64 + p) << 8);
  const float* W = (sel ? Wkc : Wkr) + ((size_t)o << 8);
  float s = sel ? bkc[o] : bkr[o];
  for (int c = 0; c < 256; ++c) s += in[c] * W[c];
  if (sel) kcol[r] = s; else krow[r] = s;
}

// ---------------- WMMA projection GEMM (16 rows x 64 cols per wave) ----------------
// PE_MODE: 0 none, 1 row-posemb, 2 col-posemb. OUTF: f32 [b,n,c]. OUTV: f16 V layout.
template <int PE_MODE, bool OUTF, bool OUTV>
__global__ void proj_gemm_kernel(const float* __restrict__ X, const float* __restrict__ PE,
                                 const _Float16* __restrict__ W, const float* __restrict__ bias,
                                 float scale, float* __restrict__ outF,
                                 _Float16* __restrict__ outV) {
  int blk = blockIdx.x;
  int ot4 = blk & 3;
  int nb = (blk >> 2) & 255;
  int b = blk >> 10;
  int lane = threadIdx.x & 31;
  int half = lane >> 4, l15 = lane & 15;
  int hoff = half ? 16 : 0;
  int n0 = nb * 16, o0 = ot4 * 64;
  int row = n0 + l15;
  const float* xr = X + (((size_t)b * NSEQ + row) << 8);
  const float* pe = nullptr;
  if (PE_MODE == 1) pe = PE + (((size_t)(b * 64 + (row & 63))) << 8);
  if (PE_MODE == 2) pe = PE + (((size_t)(b * 64 + (row >> 6))) << 8);
  v8f acc[4];
#pragma unroll
  for (int t = 0; t < 4; ++t) acc[t] = v8f_zero();
#pragma unroll
  for (int ks = 0; ks < 8; ++ks) {
    int k0 = ks * 32;
    AF A;
    load_a_f32<PE_MODE != 0>(A, xr, pe, k0, half);
#pragma unroll
    for (int t = 0; t < 4; ++t) {
      AF B;
      load_b_f16(B, W + (((size_t)(o0 + t * 16 + l15)) << 8) + k0 + hoff);
      acc[t] = wmma16(A, B, acc[t]);
    }
  }
#pragma unroll
  for (int t = 0; t < 4; ++t) {
    int ch = o0 + t * 16 + l15;
    float bch = bias[ch];
#pragma unroll
    for (int v = 0; v < 8; ++v) {
      int n = n0 + v + (half ? 8 : 0);
      float val = (acc[t][v] + bch) * scale;
      if (OUTF) outF[(((size_t)b * NSEQ + n) << 8) + ch] = val;
      if (OUTV) {
        int head = ch >> 5, d = ch & 31;
        int i = n >> 6, j = n & 63;
        // V layout: [(b*NH+head)][(j*32+d)][i]  (row-major 64-wide, TDM friendly)
        outV[((size_t)(b * NH + head) * 2048 + (j * 32 + d)) * 64 + i] = (_Float16)val;
      }
    }
  }
}

// out = A @ W^T + bias + res (rows folded over batch, 16x64 per wave)
__global__ void gemm_res_kernel(const float* __restrict__ A, const _Float16* __restrict__ W,
                                const float* __restrict__ bias, const float* __restrict__ res,
                                float* __restrict__ out) {
  int blk = blockIdx.x;
  int ot4 = blk & 3;
  int rb = blk >> 2;
  int lane = threadIdx.x & 31;
  int half = lane >> 4, l15 = lane & 15;
  int hoff = half ? 16 : 0;
  int n0 = rb * 16, o0 = ot4 * 64;
  const float* ar = A + (((size_t)(n0 + l15)) << 8);
  v8f acc[4];
#pragma unroll
  for (int t = 0; t < 4; ++t) acc[t] = v8f_zero();
#pragma unroll
  for (int ks = 0; ks < 8; ++ks) {
    int k0 = ks * 32;
    AF Af;
    load_a_f32<false>(Af, ar, nullptr, k0, half);
#pragma unroll
    for (int t = 0; t < 4; ++t) {
      AF Bf;
      load_b_f16(Bf, W + (((size_t)(o0 + t * 16 + l15)) << 8) + k0 + hoff);
      acc[t] = wmma16(Af, Bf, acc[t]);
    }
  }
#pragma unroll
  for (int t = 0; t < 4; ++t) {
    int ch = o0 + t * 16 + l15;
    float bch = bias[ch];
#pragma unroll
    for (int v = 0; v < 8; ++v) {
      size_t idx = (((size_t)(n0 + v + (half ? 8 : 0))) << 8) + ch;
      out[idx] = acc[t][v] + bch + res[idx];
    }
  }
}

// in-place layernorm, one wave per 256-wide row
__global__ void ln_kernel(float* __restrict__ y, const float* __restrict__ g,
                          const float* __restrict__ be) {
  int wave = threadIdx.x >> 5, lane = threadIdx.x & 31;
  int row = blockIdx.x * 8 + wave;
  float* yr = y + ((size_t)row << 8);
  float vals[8];
  float s = 0.f;
#pragma unroll
  for (int i = 0; i < 8; ++i) { vals[i] = yr[lane + 32 * i]; s += vals[i]; }
#pragma unroll
  for (int off = 16; off; off >>= 1) s += __shfl_xor(s, off, 32);
  float mu = s * (1.f / 256.f);
  float vs = 0.f;
#pragma unroll
  for (int i = 0; i < 8; ++i) { float d = vals[i] - mu; vs += d * d; }
#pragma unroll
  for (int off = 16; off; off >>= 1) vs += __shfl_xor(vs, off, 32);
  float rstd = rsqrtf(vs * (1.f / 256.f) + 1e-5f);
#pragma unroll
  for (int i = 0; i < 8; ++i) {
    int c = lane + 32 * i;
    yr[c] = (vals[i] - mu) * rstd * g[c] + be[c];
  }
}

// ---------------- RCDA attention core ----------------
// grid = BZ*NH*8 workgroups, 8 waves each. V_head tile (2048 x 64 f16, rows padded
// to 68 halves) staged in LDS via TDM tensor_load_to_lds when available.
__global__ void attn_kernel(const float* __restrict__ qrow, const float* __restrict__ qcol,
                            const float* __restrict__ krow, const float* __restrict__ kcol,
                            const _Float16* __restrict__ vbuf, float* __restrict__ out) {
  extern __shared__ char smem[];
  _Float16* Vlds = (_Float16*)smem;                                    // 2048*HPAD f16
  float* arow = (float*)(smem + 2048 * HPAD * 2);                      // 8 waves * 16*64 f32
  _Float16* acol = (_Float16*)(smem + 2048 * HPAD * 2 + 8 * 1024 * 4); // 8 waves * 16*64 f16

  int wg = blockIdx.x;
  int slice = wg & 7; int head = (wg >> 3) & 7; int b = wg >> 6;
  int tid = threadIdx.x, wave = tid >> 5, lane = tid & 31;
  int half = lane >> 4, l15 = lane & 15;
  int hb8 = half ? 8 : 0;
  float* myar = arow + wave * 1024;
  _Float16* myac = acol + wave * 1024;

  const _Float16* vsrc = vbuf + (((size_t)(b * NH + head)) << 17);

#if USE_TDM
  if (wave == 0) {
    unsigned long long ga = (unsigned long long)(const void*)vsrc;
    u32x4 g0; i32x8 g1; i32x4 g2, g3; i32x8 g4;
#pragma unroll
    for (int t = 0; t < 4; ++t) { g2[t] = 0; g3[t] = 0; }
#pragma unroll
    for (int t = 0; t < 8; ++t) g4[t] = 0;
    g0[0] = 1u;                                        // count=1, user descriptor
    g0[1] = 0u;                                        // lds_addr (dynamic LDS base)
    g0[2] = (unsigned)ga;                              // global_addr[31:0]
    g0[3] = (unsigned)((ga >> 32) & 0x1FFFFFFu) | (2u << 30);  // addr[56:32] | type=2
    // data_size=2B, pad_enable, pad_interval=32 dwords (code 4), pad_amount=2 dwords (code 1)
    g1[0] = (int)((1u << 16) | (1u << 20) | (4u << 22) | (1u << 25));
    g1[1] = (int)(64u << 16);     // tensor_dim0 = 64
    g1[2] = (int)(2048u << 16);   // tensor_dim1 = 2048
    g1[3] = (int)(64u << 16);     // tile_dim0 = 64
    g1[4] = (int)2048u;           // tile_dim1 = 2048, tile_dim2 = 0
    g1[5] = 64;                   // tensor_dim0_stride = 64
    g1[6] = 0; g1[7] = 0;
    __builtin_amdgcn_tensor_load_to_lds(g0, g1, g2, g3, g4, 0);
    __builtin_amdgcn_s_wait_tensorcnt(0);
  }
  __syncthreads();
#else
  for (int g = tid; g < 2048 * 16; g += 256) {
    int r = g >> 4, i0 = (g & 15) * 4;
    *(u32x2*)(Vlds + r * HPAD + i0) = *(const u32x2*)(vsrc + r * 64 + i0);
  }
  __syncthreads();
#endif

  const float* qrb = qrow + (((size_t)b) << 20) + head * 32;
  const float* qcb = qcol + (((size_t)b) << 20) + head * 32;
  const float* krb = krow + (((size_t)(b * 64)) << 8) + head * 32;
  const float* kcb = kcol + (((size_t)(b * 64)) << 8) + head * 32;

  for (int it = 0; it < 4; ++it) {
    int qb = slice * 32 + it * 8 + wave;
    int n0 = qb * 16;

    // ---- column logits -> myar (f32), softmax -> myac (f16) ----
    {
      AF qA;
      load_a_f32<false>(qA, qcb + (((size_t)(n0 + l15)) << 8), nullptr, 0, half);
#pragma unroll
      for (int t = 0; t < 4; ++t) {
        AF Bf;
        load_b_f32cvt(Bf, kcb + (((size_t)(t * 16 + l15)) << 8) + (half ? 16 : 0));
        v8f lg = wmma16(qA, Bf, v8f_zero());
#pragma unroll
        for (int v = 0; v < 8; ++v) myar[(v + hb8) * 64 + t * 16 + l15] = lg[v];
      }
    }
    if (lane < 16) {
      float mx = -1e30f;
      for (int w = 0; w < 64; ++w) mx = fmaxf(mx, myar[lane * 64 + w]);
      float s = 0.f;
      for (int w = 0; w < 64; ++w) {
        float e = __expf(myar[lane * 64 + w] - mx);
        myar[lane * 64 + w] = e; s += e;
      }
      float inv = 1.f / s;
      for (int w = 0; w < 64; ++w)
        myac[lane * 64 + w] = (_Float16)(myar[lane * 64 + w] * inv);
    }

    // ---- row logits -> myar (f32), softmax in place ----
    {
      AF qA;
      load_a_f32<false>(qA, qrb + (((size_t)(n0 + l15)) << 8), nullptr, 0, half);
#pragma unroll
      for (int t = 0; t < 4; ++t) {
        AF Bf;
        load_b_f32cvt(Bf, krb + (((size_t)(t * 16 + l15)) << 8) + (half ? 16 : 0));
        v8f lg = wmma16(qA, Bf, v8f_zero());
#pragma unroll
        for (int v = 0; v < 8; ++v) myar[(v + hb8) * 64 + t * 16 + l15] = lg[v];
      }
    }
    if (lane < 16) {
      float mx = -1e30f;
      for (int w = 0; w < 64; ++w) mx = fmaxf(mx, myar[lane * 64 + w]);
      float s = 0.f;
      for (int w = 0; w < 64; ++w) {
        float e = __expf(myar[lane * 64 + w] - mx);
        myar[lane * 64 + w] = e; s += e;
      }
      float inv = 1.f / s;
      for (int w = 0; w < 64; ++w) myar[lane * 64 + w] *= inv;
    }

    // ---- attn_col A-fragments (K = h = 64 -> 2 ksteps), reused for all w ----
    AF cA0, cA1;
    {
      const _Float16* tr = myac + l15 * 64;
      cA0.q[0] = *(const u32x4*)(tr + (half ? 16 : 0));
      cA0.q[1] = *(const u32x4*)(tr + (half ? 16 : 0) + 8);
      cA1.q[0] = *(const u32x4*)(tr + 32 + (half ? 16 : 0));
      cA1.q[1] = *(const u32x4*)(tr + 32 + (half ? 16 : 0) + 8);
    }

    // ---- main loop: tmp = ac @ V[:,w,:] (WMMA over h), fused row reduction ----
    v8f acc0 = v8f_zero(), acc1 = v8f_zero();
    int kb0 = half ? 16 : 0;
    for (int w = 0; w < 64; ++w) {
      const _Float16* vb0 = Vlds + (w * 32 + l15) * HPAD + kb0;        // tile d=l15
      const _Float16* vb1 = Vlds + (w * 32 + 16 + l15) * HPAD + kb0;   // tile d=16+l15
      AF B;
      v8f t0, t1;
#pragma unroll
      for (int j = 0; j < 4; ++j) B.p[j] = *(const u32x2*)(vb0 + 4 * j);
      t0 = wmma16(cA0, B, v8f_zero());
#pragma unroll
      for (int j = 0; j < 4; ++j) B.p[j] = *(const u32x2*)(vb0 + 32 + 4 * j);
      t0 = wmma16(cA1, B, t0);
#pragma unroll
      for (int j = 0; j < 4; ++j) B.p[j] = *(const u32x2*)(vb1 + 4 * j);
      t1 = wmma16(cA0, B, v8f_zero());
#pragma unroll
      for (int j = 0; j < 4; ++j) B.p[j] = *(const u32x2*)(vb1 + 32 + 4 * j);
      t1 = wmma16(cA1, B, t1);
#pragma unroll
      for (int v = 0; v < 8; ++v) {
        float a = myar[(v + hb8) * 64 + w];
        acc0[v] += a * t0[v];
        acc1[v] += a * t1[v];
      }
    }

    // ---- store [b, n, head*32 + d] ----
#pragma unroll
    for (int v = 0; v < 8; ++v) {
      int n = n0 + v + hb8;
      float* o = out + (((size_t)b * NSEQ + n) << 8) + head * 32;
      o[l15] = acc0[v];
      o[16 + l15] = acc1[v];
    }
  }
}

// ---------------- fused FFN + LN2 + output transpose ----------------
__global__ void ffn_kernel(const float* __restrict__ y, const _Float16* __restrict__ W1,
                           const float* __restrict__ b1, const _Float16* __restrict__ W2,
                           const float* __restrict__ b2, const float* __restrict__ g2,
                           const float* __restrict__ be2, float* __restrict__ out) {
  __shared__ alignas(16) _Float16 tlds[16 * FF];
  __shared__ float slds[16 * CDIM];
  __shared__ float mu[16], rs[16];
  int n0g = blockIdx.x * 16;
  int tid = threadIdx.x, wave = tid >> 5, lane = tid & 31;
  int half = lane >> 4, l15 = lane & 15;
  int hb8 = half ? 8 : 0;
  int hoff = half ? 16 : 0;
  const float* yr = y + (((size_t)(n0g + l15)) << 8);

  // Phase A: t = relu(y @ W1^T + b1), this wave covers cols [wave*128, wave*128+128)
  v8f acc[8];
#pragma unroll
  for (int t = 0; t < 8; ++t) acc[t] = v8f_zero();
  for (int ks = 0; ks < 8; ++ks) {
    int k0 = ks * 32;
    AF A;
    load_a_f32<false>(A, yr, nullptr, k0, half);
#pragma unroll
    for (int t = 0; t < 8; ++t) {
      int o = wave * 128 + t * 16 + l15;
      AF B;
      load_b_f16(B, W1 + (((size_t)o) << 8) + k0 + hoff);
      acc[t] = wmma16(A, B, acc[t]);
    }
  }
#pragma unroll
  for (int t = 0; t < 8; ++t) {
    int c = wave * 128 + t * 16 + l15;
    float bc = b1[c];
#pragma unroll
    for (int v = 0; v < 8; ++v) {
      float val = acc[t][v] + bc;
      tlds[(v + hb8) * FF + c] = (_Float16)fmaxf(val, 0.f);
    }
  }
  __syncthreads();

  // Phase B: out = t @ W2^T + b2 + y, this wave covers cols [wave*32, wave*32+32)
  v8f o0 = v8f_zero(), o1 = v8f_zero();
  int c0 = wave * 32, c1 = wave * 32 + 16;
  for (int ks = 0; ks < 32; ++ks) {
    int k0 = ks * 32;
    AF A;
    {
      const _Float16* tr = tlds + l15 * FF + k0 + hoff;
      A.q[0] = *(const u32x4*)(tr);
      A.q[1] = *(const u32x4*)(tr + 8);
    }
    AF B;
    load_b_f16(B, W2 + (((size_t)(c0 + l15)) << 10) + k0 + hoff);
    o0 = wmma16(A, B, o0);
    load_b_f16(B, W2 + (((size_t)(c1 + l15)) << 10) + k0 + hoff);
    o1 = wmma16(A, B, o1);
  }
#pragma unroll
  for (int v = 0; v < 8; ++v) {
    int m = v + hb8;
    size_t ybase = ((size_t)(n0g + m)) << 8;
    slds[m * CDIM + c0 + l15] = o0[v] + b2[c0 + l15] + y[ybase + c0 + l15];
    slds[m * CDIM + c1 + l15] = o1[v] + b2[c1 + l15] + y[ybase + c1 + l15];
  }
  __syncthreads();

  // Phase C: LN2 + NCHW store
  if (tid < 16) {
    float s = 0.f;
    for (int c = 0; c < CDIM; ++c) s += slds[tid * CDIM + c];
    float m_ = s * (1.f / 256.f);
    float vs = 0.f;
    for (int c = 0; c < CDIM; ++c) { float d = slds[tid * CDIM + c] - m_; vs += d * d; }
    mu[tid] = m_;
    rs[tid] = rsqrtf(vs * (1.f / 256.f) + 1e-5f);
  }
  __syncthreads();
  for (int e = tid; e < 16 * CDIM; e += 256) {
    int m = e >> 8, c = e & 255;
    int gn = n0g + m; int b = gn >> 12; int n = gn & 4095;
    float val = (slds[e] - mu[m]) * rs[m] * g2[c] + be2[c];
    out[(((size_t)(b * CDIM + c)) << 12) + n] = val;
  }
}

// ---------------- workspace layout ----------------
static constexpr size_t OFF_X    = 0;
static constexpr size_t OFF_QROW = OFF_X + 8388608;      // aliased as ybuf after attention
static constexpr size_t OFF_QCOL = OFF_QROW + 8388608;
static constexpr size_t OFF_ATTN = OFF_QCOL + 8388608;
static constexpr size_t OFF_V    = OFF_ATTN + 8388608;   // f16
static constexpr size_t OFF_KROW = OFF_V + 4194304;
static constexpr size_t OFF_KCOL = OFF_KROW + 131072;
static constexpr size_t OFF_XRM  = OFF_KCOL + 131072;
static constexpr size_t OFF_XCM  = OFF_XRM + 131072;
static constexpr size_t OFF_WQR  = OFF_XCM + 131072;
static constexpr size_t OFF_WQC  = OFF_WQR + 131072;
static constexpr size_t OFF_WV   = OFF_WQC + 131072;
static constexpr size_t OFF_WO   = OFF_WV + 131072;
static constexpr size_t OFF_W1   = OFF_WO + 131072;
static constexpr size_t OFF_W2   = OFF_W1 + 524288;

extern "C" void kernel_launch(void* const* d_in, const int* in_sizes, int n_in,
                              void* d_out, int out_size, void* d_ws, size_t ws_size,
                              hipStream_t stream) {
  (void)in_sizes; (void)n_in; (void)out_size; (void)ws_size;
  const float* src        = (const float*)d_in[0];
  const float* posemb_row = (const float*)d_in[1];
  const float* posemb_col = (const float*)d_in[2];
  const float* Wq_row = (const float*)d_in[3];  const float* bq_row = (const float*)d_in[4];
  const float* Wq_col = (const float*)d_in[5];  const float* bq_col = (const float*)d_in[6];
  const float* Wk_row = (const float*)d_in[7];  const float* bk_row = (const float*)d_in[8];
  const float* Wk_col = (const float*)d_in[9];  const float* bk_col = (const float*)d_in[10];
  const float* Wv = (const float*)d_in[11];     const float* bv = (const float*)d_in[12];
  const float* Wo = (const float*)d_in[13];     const float* bo = (const float*)d_in[14];
  const float* g1 = (const float*)d_in[15];     const float* be1 = (const float*)d_in[16];
  const float* W1 = (const float*)d_in[17];     const float* b1 = (const float*)d_in[18];
  const float* W2 = (const float*)d_in[19];     const float* b2 = (const float*)d_in[20];
  const float* g2 = (const float*)d_in[21];     const float* be2 = (const float*)d_in[22];

  char* ws = (char*)d_ws;
  float* xbuf  = (float*)(ws + OFF_X);
  float* qrowb = (float*)(ws + OFF_QROW);
  float* qcolb = (float*)(ws + OFF_QCOL);
  float* attnb = (float*)(ws + OFF_ATTN);
  float* ybuf  = (float*)(ws + OFF_QROW);   // alias: q_row dead after attention
  _Float16* vbufh = (_Float16*)(ws + OFF_V);
  float* krowb = (float*)(ws + OFF_KROW);
  float* kcolb = (float*)(ws + OFF_KCOL);
  float* xrm   = (float*)(ws + OFF_XRM);
  float* xcm   = (float*)(ws + OFF_XCM);
  _Float16* wqr = (_Float16*)(ws + OFF_WQR);
  _Float16* wqc = (_Float16*)(ws + OFF_WQC);
  _Float16* wvh = (_Float16*)(ws + OFF_WV);
  _Float16* woh = (_Float16*)(ws + OFF_WO);
  _Float16* w1h = (_Float16*)(ws + OFF_W1);
  _Float16* w2h = (_Float16*)(ws + OFF_W2);

  // weight conversion to f16
  cvt_f16_kernel<<<256, 256, 0, stream>>>(Wq_row, wqr, CDIM * CDIM);
  cvt_f16_kernel<<<256, 256, 0, stream>>>(Wq_col, wqc, CDIM * CDIM);
  cvt_f16_kernel<<<256, 256, 0, stream>>>(Wv, wvh, CDIM * CDIM);
  cvt_f16_kernel<<<256, 256, 0, stream>>>(Wo, woh, CDIM * CDIM);
  cvt_f16_kernel<<<1024, 256, 0, stream>>>(W1, w1h, FF * CDIM);
  cvt_f16_kernel<<<1024, 256, 0, stream>>>(W2, w2h, CDIM * FF);

  // NCHW -> [b,n,c]
  transpose_src_kernel<<<(BZ * NSEQ * CDIM + 255) / 256, 256, 0, stream>>>(src, xbuf);

  // axis means (+posemb) and decoupled-key projections
  mean_pe_kernel<<<(2 * BZ * 64 * CDIM + 255) / 256, 256, 0, stream>>>(
      xbuf, posemb_row, posemb_col, xrm, xcm);
  kproj_kernel<<<(2 * BZ * 64 * CDIM + 255) / 256, 256, 0, stream>>>(
      xrm, xcm, Wk_row, bk_row, Wk_col, bk_col, krowb, kcolb);

  // WMMA projections
  const float qscale = 0.17677669529663687f;  // 32^-0.5
  proj_gemm_kernel<1, true, false><<<BZ * 256 * 4, 32, 0, stream>>>(
      xbuf, posemb_row, wqr, bq_row, qscale, qrowb, (_Float16*)nullptr);
  proj_gemm_kernel<2, true, false><<<BZ * 256 * 4, 32, 0, stream>>>(
      xbuf, posemb_col, wqc, bq_col, qscale, qcolb, (_Float16*)nullptr);
  proj_gemm_kernel<0, false, true><<<BZ * 256 * 4, 32, 0, stream>>>(
      xbuf, (const float*)nullptr, wvh, bv, 1.0f, (float*)nullptr, vbufh);

  // RCDA attention core (exactly 320KB LDS: 278528 V tile + softmax scratch)
  size_t attn_lds = 2048 * HPAD * 2 + 8 * 1024 * 4 + 8 * 1024 * 2;  // 327680 B
  (void)hipFuncSetAttribute(reinterpret_cast<const void*>(&attn_kernel),
                            hipFuncAttributeMaxDynamicSharedMemorySize, (int)attn_lds);
  attn_kernel<<<BZ * NH * 8, 256, attn_lds, stream>>>(qrowb, qcolb, krowb, kcolb, vbufh, attnb);

  // Wo projection + residual, then LN1 in place
  gemm_res_kernel<<<(BZ * NSEQ / 16) * 4, 32, 0, stream>>>(attnb, woh, bo, xbuf, ybuf);
  ln_kernel<<<BZ * NSEQ / 8, 256, 0, stream>>>(ybuf, g1, be1);

  // fused FFN + LN2 + NCHW output
  ffn_kernel<<<BZ * NSEQ / 16, 256, 0, stream>>>(ybuf, w1h, b1, w2h, b2, g2, be2,
                                                 (float*)d_out);
}
